// CausalSelfAttention_11759620456818
// MI455X (gfx1250) — matrix operand order repaired and checked by reading, not compiled, not run on hardware
//
#include <hip/hip_runtime.h>

typedef __attribute__((ext_vector_type(16))) _Float16 v16h;
typedef __attribute__((ext_vector_type(8)))  _Float16 v8h;
typedef __attribute__((ext_vector_type(8)))  float    v8f;

#define NBATCH 4
#define TSEQ   2048
#define EMB    1024
#define NHEAD  16
#define HDIM   64

// ---------------- CDNA5 async copy helpers (gfx1250) ---------------------
// GLOBAL_LOAD_ASYNC_TO_LDS_B128: memory -> LDS without touching VGPR data,
// tracked by ASYNCcnt (CDNA5 ISA).
__device__ __forceinline__ void async_ld_b128(void* lds, const void* g) {
  asm volatile("global_load_async_to_lds_b128 %0, %1, off"
               :
               : "v"((unsigned)(unsigned long long)lds),
                 "v"((unsigned long long)g)
               : "memory");
}
__device__ __forceinline__ void wait_async0() {
  asm volatile("s_wait_asynccnt 0x0" ::: "memory");
}

// ---- WMMA fragment helpers (CDNA5 16x16x32 f16 layouts, ISA 7.12.2) ----
// A: lane holds row M=lane&15; halfs [0..7] at K-offset (lane>>4)*8, halfs [8..15] at +16.
__device__ __forceinline__ v16h frag_a(const _Float16* row, int kof) {
  v8h lo = *(const v8h*)(row + kof);
  v8h hi = *(const v8h*)(row + kof + 16);
  v16h r;
#pragma unroll
  for (int i = 0; i < 8; ++i) { r[i] = lo[i]; r[i + 8] = hi[i]; }
  return r;
}
// B: lane holds col N=lane&15; halfs [0..7] at K-offset (lane>>4)*8, halfs [8..15] at +16.
__device__ __forceinline__ v16h frag_b(const _Float16* col) {
  v8h lo = *(const v8h*)(col);
  v8h hi = *(const v8h*)(col + 16);
  v16h r;
#pragma unroll
  for (int i = 0; i < 8; ++i) { r[i] = lo[i]; r[i + 8] = hi[i]; }
  return r;
}

__device__ __forceinline__ float rowmax16(float v) {
#pragma unroll
  for (int m = 1; m <= 8; m <<= 1) v = fmaxf(v, __shfl_xor(v, m, 32));
  return v;
}
__device__ __forceinline__ float rowsum16(float v) {
#pragma unroll
  for (int m = 1; m <= 8; m <<= 1) v += __shfl_xor(v, m, 32);
  return v;
}

// ======================= Kernel 1: QKV projection ========================
// C[8192,1024] = X[8192,1024](f32) * W[1024,1024](f32) + b
// 128x128 tile per 256-thread block; f16 staged in LDS; f32 accumulate.
__global__ __launch_bounds__(256)
void qkv_proj_kernel(const float* __restrict__ x, const float* __restrict__ W,
                     const float* __restrict__ bias,
                     _Float16* __restrict__ dst16, float* __restrict__ dst32,
                     float qscale) {
  __shared__ __align__(16) _Float16 Xs[128][40];  // [row][k]
  __shared__ __align__(16) _Float16 Wt[128][40];  // [col][k] (transposed)

  const int tid  = threadIdx.x;
  const int lane = tid & 31;
  const int wave = tid >> 5;
  const int mt = blockIdx.x & 63;
  const int nt = blockIdx.x >> 6;
  const int rowbase = mt * 128;
  const int colbase = nt * 128;

  const int wmo = (wave & 3) * 32;
  const int wno = (wave >> 2) * 64;
  const int fr  = lane & 15;
  const int kofA = (lane >> 4) * 8;
  const int kofB = (lane >> 4) * 8;

  v8f acc[2][4];
#pragma unroll
  for (int mi = 0; mi < 2; ++mi)
#pragma unroll
    for (int ni = 0; ni < 4; ++ni)
#pragma unroll
      for (int j = 0; j < 8; ++j) acc[mi][ni][j] = 0.0f;

  const int xr = tid >> 1;           // 0..127
  const int xc = (tid & 1) * 16;     // 0 / 16
  const int wk = tid >> 3;           // 0..31
  const int wn = (tid & 7) * 16;     // 0..112

  for (int kk = 0; kk < EMB; kk += 32) {
    // stage X tile (convert f32->f16)
    const float4* xp = (const float4*)(x + (size_t)(rowbase + xr) * EMB + kk + xc);
#pragma unroll
    for (int i = 0; i < 4; ++i) {
      float4 f = xp[i];
      Xs[xr][xc + i * 4 + 0] = (_Float16)f.x;
      Xs[xr][xc + i * 4 + 1] = (_Float16)f.y;
      Xs[xr][xc + i * 4 + 2] = (_Float16)f.z;
      Xs[xr][xc + i * 4 + 3] = (_Float16)f.w;
    }
    // stage W tile transposed (convert f32->f16)
    const float4* wp = (const float4*)(W + (size_t)(kk + wk) * EMB + colbase + wn);
#pragma unroll
    for (int i = 0; i < 4; ++i) {
      float4 f = wp[i];
      Wt[wn + i * 4 + 0][wk] = (_Float16)f.x;
      Wt[wn + i * 4 + 1][wk] = (_Float16)f.y;
      Wt[wn + i * 4 + 2][wk] = (_Float16)f.z;
      Wt[wn + i * 4 + 3][wk] = (_Float16)f.w;
    }
    // prefetch next K-chunk into cache while we compute this one
    if (kk + 32 < EMB) {
      __builtin_prefetch(x + (size_t)(rowbase + xr) * EMB + kk + 32 + xc, 0, 0);
      __builtin_prefetch(W + (size_t)(kk + 32 + wk) * EMB + colbase + wn, 0, 0);
    }
    __syncthreads();

    v16h a[2], b[4];
#pragma unroll
    for (int mi = 0; mi < 2; ++mi) a[mi] = frag_a(&Xs[wmo + mi * 16 + fr][0], kofA);
#pragma unroll
    for (int ni = 0; ni < 4; ++ni) b[ni] = frag_b(&Wt[wno + ni * 16 + fr][kofB]);
#pragma unroll
    for (int mi = 0; mi < 2; ++mi)
#pragma unroll
      for (int ni = 0; ni < 4; ++ni)
        acc[mi][ni] = __builtin_amdgcn_wmma_f32_16x16x32_f16(
            false, a[mi], false, b[ni], (short)0, acc[mi][ni], false, false);
    __syncthreads();
  }

  const int rhalf = (lane >> 4) * 8;
  if (dst32) {  // K / V: also emit fp32 "present" (uniform branch)
#pragma unroll
    for (int mi = 0; mi < 2; ++mi)
#pragma unroll
      for (int ni = 0; ni < 4; ++ni) {
        const int col = colbase + wno + ni * 16 + fr;
        const float bval = bias[col];
        const int h = col >> 6, d = col & 63;
#pragma unroll
        for (int j = 0; j < 8; ++j) {
          const int r = rowbase + wmo + mi * 16 + j + rhalf;
          const int n = r >> 11, t = r & 2047;
          const float val = acc[mi][ni][j] + bval;
          const size_t idx = ((size_t)((n * NHEAD + h) * TSEQ + t)) * HDIM + d;
          dst16[idx] = (_Float16)val;
          dst32[idx] = val;  // present[kv] layout == [n][h][t][d]
        }
      }
  } else {      // Q: f16 only, pre-scaled by 1/sqrt(hd)
#pragma unroll
    for (int mi = 0; mi < 2; ++mi)
#pragma unroll
      for (int ni = 0; ni < 4; ++ni) {
        const int col = colbase + wno + ni * 16 + fr;
        const float bval = bias[col];
        const int h = col >> 6, d = col & 63;
#pragma unroll
        for (int j = 0; j < 8; ++j) {
          const int r = rowbase + wmo + mi * 16 + j + rhalf;
          const int n = r >> 11, t = r & 2047;
          const float val = (acc[mi][ni][j] + bval) * qscale;
          const size_t idx = ((size_t)((n * NHEAD + h) * TSEQ + t)) * HDIM + d;
          dst16[idx] = (_Float16)val;
        }
      }
  }
}

// ======================= Kernel 2: flash attention =======================
// One block per (n, h, 128-query tile). 64-key blocks, online softmax.
__global__ __launch_bounds__(256)
void flash_attn_kernel(const _Float16* __restrict__ qf,
                       const _Float16* __restrict__ kf,
                       const _Float16* __restrict__ vf,
                       _Float16* __restrict__ ob) {
  __shared__ __align__(16) _Float16 Kt[64][72];      // [key][hd]
  __shared__ __align__(16) _Float16 Vt[64][72];      // [hd][key] (transposed)
  __shared__ __align__(16) _Float16 Ps[8][16][72];   // per-wave P tile

  const int tid  = threadIdx.x;
  const int lane = tid & 31;
  const int wave = tid >> 5;
  const int nh = blockIdx.x >> 4;           // n*16+h
  const int qb = blockIdx.x & 15;
  const int n = nh >> 4, h = nh & 15;
  const int qw = qb * 128 + wave * 16;      // wave's first query row
  const int fr = lane & 15;
  const int kofA = (lane >> 4) * 8;
  const int kofB = (lane >> 4) * 8;
  const int rhalf = (lane >> 4) * 8;

  // Q fragments (Q pre-scaled by 1/sqrt(hd) at projection)
  const _Float16* qrow = qf + ((size_t)nh * TSEQ + qw + fr) * HDIM;
  v16h aq[2];
#pragma unroll
  for (int c = 0; c < 2; ++c) aq[c] = frag_a(qrow + c * 32, kofA);

  float mstate[8], lstate[8];
#pragma unroll
  for (int j = 0; j < 8; ++j) { mstate[j] = -INFINITY; lstate[j] = 0.0f; }
  v8f acc_o[4];
#pragma unroll
  for (int di = 0; di < 4; ++di)
#pragma unroll
    for (int j = 0; j < 8; ++j) acc_o[di][j] = 0.0f;

  const int lr = tid >> 2;          // 0..63
  const int lc = (tid & 3) * 16;    // 0..48

  const int nblocks = 2 * qb + 2;
  for (int kb = 0; kb < nblocks; ++kb) {
    // K block: pure f16 copy -> CDNA5 async global->LDS path (ASYNCcnt)
    const _Float16* kg = kf + ((size_t)nh * TSEQ + kb * 64 + lr) * HDIM + lc;
    async_ld_b128(&Kt[lr][lc],     kg);
    async_ld_b128(&Kt[lr][lc + 8], kg + 8);
    // V block needs a transpose -> stage through VGPRs
    const _Float16* vg = vf + ((size_t)nh * TSEQ + kb * 64 + lr) * HDIM + lc;
    v8h v0 = *(const v8h*)vg;
    v8h v1 = *(const v8h*)(vg + 8);
#pragma unroll
    for (int i = 0; i < 8; ++i) {
      Vt[lc + i][lr]     = v0[i];
      Vt[lc + 8 + i][lr] = v1[i];
    }
    wait_async0();
    __syncthreads();

    // S = Q @ K^T   (16 x 64 per wave)
    v8f s[4];
#pragma unroll
    for (int ni = 0; ni < 4; ++ni)
#pragma unroll
      for (int j = 0; j < 8; ++j) s[ni][j] = 0.0f;
#pragma unroll
    for (int c = 0; c < 2; ++c)
#pragma unroll
      for (int ni = 0; ni < 4; ++ni) {
        v16h b = frag_b(&Kt[ni * 16 + fr][c * 32 + kofB]);
        s[ni] = __builtin_amdgcn_wmma_f32_16x16x32_f16(
            false, aq[c], false, b, (short)0, s[ni], false, false);
      }

    // causal mask (only the two diagonal blocks need it)
    if (kb >= 2 * qb) {
#pragma unroll
      for (int ni = 0; ni < 4; ++ni) {
        const int key = kb * 64 + ni * 16 + fr;
#pragma unroll
        for (int j = 0; j < 8; ++j) {
          const int q = qw + j + rhalf;
          if (key > q) s[ni][j] = -INFINITY;
        }
      }
    }

    // online softmax update (rows split across 16-lane halves per C layout)
#pragma unroll
    for (int j = 0; j < 8; ++j) {
      float rm = fmaxf(fmaxf(s[0][j], s[1][j]), fmaxf(s[2][j], s[3][j]));
      rm = rowmax16(rm);
      const float mnew = fmaxf(mstate[j], rm);
      const float corr = __expf(mstate[j] - mnew);
      mstate[j] = mnew;
      float p0 = __expf(s[0][j] - mnew);
      float p1 = __expf(s[1][j] - mnew);
      float p2 = __expf(s[2][j] - mnew);
      float p3 = __expf(s[3][j] - mnew);
      s[0][j] = p0; s[1][j] = p1; s[2][j] = p2; s[3][j] = p3;
      lstate[j] = lstate[j] * corr + rowsum16(p0 + p1 + p2 + p3);
#pragma unroll
      for (int di = 0; di < 4; ++di) acc_o[di][j] *= corr;
    }

    // P -> per-wave LDS tile (re-fragment as WMMA-A), then O += P @ V
#pragma unroll
    for (int ni = 0; ni < 4; ++ni)
#pragma unroll
      for (int j = 0; j < 8; ++j)
        Ps[wave][j + rhalf][ni * 16 + fr] = (_Float16)s[ni][j];
    // per-wave buffer: wave-local LDS ordering suffices, no barrier needed
#pragma unroll
    for (int c = 0; c < 2; ++c) {
      v16h ap = frag_a(&Ps[wave][fr][0], c * 32 + kofA);
#pragma unroll
      for (int di = 0; di < 4; ++di) {
        v16h bv = frag_b(&Vt[di * 16 + fr][c * 32 + kofB]);
        acc_o[di] = __builtin_amdgcn_wmma_f32_16x16x32_f16(
            false, ap, false, bv, (short)0, acc_o[di], false, false);
      }
    }
    __syncthreads();
  }

  // normalize and write attention output as f16 [n*T + q][h*64 + d]
#pragma unroll
  for (int j = 0; j < 8; ++j) {
    const float inv = 1.0f / lstate[j];
    const int q = qw + j + rhalf;
#pragma unroll
    for (int di = 0; di < 4; ++di) {
      const int d = di * 16 + fr;
      ob[((size_t)(n * TSEQ + q)) * EMB + h * HDIM + d] =
          (_Float16)(acc_o[di][j] * inv);
    }
  }
}

// ===================== Kernel 3: output projection =======================
// y[8192,1024](f32) = A[8192,1024](f16) * Wp(f32->f16) + bp
__global__ __launch_bounds__(256)
void out_proj_kernel(const _Float16* __restrict__ A, const float* __restrict__ W,
                     const float* __restrict__ bias, float* __restrict__ y) {
  __shared__ __align__(16) _Float16 As[128][40];
  __shared__ __align__(16) _Float16 Wt[128][40];

  const int tid  = threadIdx.x;
  const int lane = tid & 31;
  const int wave = tid >> 5;
  const int mt = blockIdx.x & 63;
  const int nt = blockIdx.x >> 6;
  const int rowbase = mt * 128;
  const int colbase = nt * 128;

  const int wmo = (wave & 3) * 32;
  const int wno = (wave >> 2) * 64;
  const int fr  = lane & 15;
  const int kofA = (lane >> 4) * 8;
  const int kofB = (lane >> 4) * 8;

  v8f acc[2][4];
#pragma unroll
  for (int mi = 0; mi < 2; ++mi)
#pragma unroll
    for (int ni = 0; ni < 4; ++ni)
#pragma unroll
      for (int j = 0; j < 8; ++j) acc[mi][ni][j] = 0.0f;

  const int ar = tid >> 1;
  const int ac = (tid & 1) * 16;
  const int wk = tid >> 3;
  const int wn = (tid & 7) * 16;

  for (int kk = 0; kk < EMB; kk += 32) {
    // A tile is already f16 in the right layout: async global->LDS copy
    const _Float16* ag = A + (size_t)(rowbase + ar) * EMB + kk + ac;
    async_ld_b128(&As[ar][ac],     ag);
    async_ld_b128(&As[ar][ac + 8], ag + 8);
    // W tile: convert f32->f16 + transpose through VGPRs
    const float4* wp = (const float4*)(W + (size_t)(kk + wk) * EMB + colbase + wn);
#pragma unroll
    for (int i = 0; i < 4; ++i) {
      float4 f = wp[i];
      Wt[wn + i * 4 + 0][wk] = (_Float16)f.x;
      Wt[wn + i * 4 + 1][wk] = (_Float16)f.y;
      Wt[wn + i * 4 + 2][wk] = (_Float16)f.z;
      Wt[wn + i * 4 + 3][wk] = (_Float16)f.w;
    }
    if (kk + 32 < EMB) {
      __builtin_prefetch(A + (size_t)(rowbase + ar) * EMB + kk + 32 + ac, 0, 0);
      __builtin_prefetch(W + (size_t)(kk + 32 + wk) * EMB + colbase + wn, 0, 0);
    }
    wait_async0();
    __syncthreads();

    v16h a[2], b[4];
#pragma unroll
    for (int mi = 0; mi < 2; ++mi) a[mi] = frag_a(&As[wmo + mi * 16 + fr][0], kofA);
#pragma unroll
    for (int ni = 0; ni < 4; ++ni) b[ni] = frag_b(&Wt[wno + ni * 16 + fr][kofB]);
#pragma unroll
    for (int mi = 0; mi < 2; ++mi)
#pragma unroll
      for (int ni = 0; ni < 4; ++ni)
        acc[mi][ni] = __builtin_amdgcn_wmma_f32_16x16x32_f16(
            false, a[mi], false, b[ni], (short)0, acc[mi][ni], false, false);
    __syncthreads();
  }

  const int rhalf = (lane >> 4) * 8;
#pragma unroll
  for (int mi = 0; mi < 2; ++mi)
#pragma unroll
    for (int ni = 0; ni < 4; ++ni) {
      const int col = colbase + wno + ni * 16 + fr;
      const float bval = bias[col];
#pragma unroll
      for (int j = 0; j < 8; ++j) {
        const int r = rowbase + wmo + mi * 16 + j + rhalf;
        y[(size_t)r * EMB + col] = acc[mi][ni][j] + bval;
      }
    }
}

// ============================== launcher =================================
extern "C" void kernel_launch(void* const* d_in, const int* in_sizes, int n_in,
                              void* d_out, int out_size, void* d_ws, size_t ws_size,
                              hipStream_t stream) {
  (void)in_sizes; (void)n_in; (void)out_size; (void)ws_size;
  const float* x  = (const float*)d_in[0];
  // d_in[1] = mask (causal mask is hardcoded)
  const float* Wq = (const float*)d_in[2];
  const float* bq = (const float*)d_in[3];
  const float* Wk = (const float*)d_in[4];
  const float* bk = (const float*)d_in[5];
  const float* Wv = (const float*)d_in[6];
  const float* bv = (const float*)d_in[7];
  const float* Wp = (const float*)d_in[8];
  const float* bp = (const float*)d_in[9];

  float* y  = (float*)d_out;                                    // [N,T,E]
  float* pk = y + (size_t)NBATCH * TSEQ * EMB;                  // present[0]
  float* pv = pk + (size_t)NBATCH * NHEAD * TSEQ * HDIM;        // present[1]

  _Float16* qf = (_Float16*)d_ws;                               // 16 MB each
  _Float16* kf = qf + (size_t)NBATCH * TSEQ * EMB;
  _Float16* vf = kf + (size_t)NBATCH * TSEQ * EMB;
  _Float16* ob = vf + (size_t)NBATCH * TSEQ * EMB;

  const dim3 blk(256);
  const dim3 gGemm(64 * 8);           // 64 M-tiles x 8 N-tiles
  const dim3 gAttn(NBATCH * NHEAD * (TSEQ / 128));

  qkv_proj_kernel<<<gGemm, blk, 0, stream>>>(x, Wq, bq, qf, nullptr, 0.125f);
  qkv_proj_kernel<<<gGemm, blk, 0, stream>>>(x, Wk, bk, kf, pk, 1.0f);
  qkv_proj_kernel<<<gGemm, blk, 0, stream>>>(x, Wv, bv, vf, pv, 1.0f);
  flash_attn_kernel<<<gAttn, blk, 0, stream>>>(qf, kf, vf, ob);
  out_proj_kernel<<<gGemm, blk, 0, stream>>>(ob, Wp, bp, y);
}
